// RNAEmbedding_33148557591016
// MI455X (gfx1250) — matrix operand order repaired and checked by reading, not compiled, not run on hardware
//
#include <hip/hip_runtime.h>
#include <hip/hip_bf16.h>

typedef __attribute__((ext_vector_type(16))) _Float16 v16h;
typedef __attribute__((ext_vector_type(8)))  _Float16 v8h;
typedef __attribute__((ext_vector_type(8)))  float    v8f;

#define RADIUS    2
#define EMBED     64
#define NGENE_PAD 32          // pad gene axis to WMMA K=32
#define H_DIM     512         // fixed by the reference
#define W_DIM     512
#define HW        (H_DIM * W_DIM)
#define PIX_TILES 4           // 16-pixel tiles per wave (64 px/wave)

// ---------------------------------------------------------------------------
// Pass 0: zero the per-pixel winning-gene map (d_ws is poisoned by harness).
// ---------------------------------------------------------------------------
__global__ void zero_gmap(int4* __restrict__ gmap4, int n4) {
    int i = blockIdx.x * blockDim.x + threadIdx.x;
    int stride = gridDim.x * blockDim.x;
    int4 z = {0, 0, 0, 0};
    for (; i < n4; i += stride) gmap4[i] = z;
}

// ---------------------------------------------------------------------------
// Pass 1: scatter the winning gene index into the 5x5 clamped window.
// 4-byte proxy scatter (L2-resident) instead of 256-byte embedding scatter.
// Race winner is arbitrary, matching JAX's unspecified duplicate-.set order.
// blockIdx.y = batch  ->  no integer divide, scalar base addresses.
// ---------------------------------------------------------------------------
__global__ void scatter_genes(const int* __restrict__ gene,
                              const int* __restrict__ coord,
                              int* __restrict__ gmap,
                              int N) {
    int i = blockIdx.x * blockDim.x + threadIdx.x;
    if (i >= N) return;
    int b = blockIdx.y;
    const int* geneB  = gene  + (size_t)b * N;
    const int* coordB = coord + (size_t)b * N * 2;
    int* gmapB = gmap + (size_t)b * HW;

    int g = geneB[i];
    int y = coordB[2 * i + 0];
    int x = coordB[2 * i + 1];
#pragma unroll
    for (int oy = -RADIUS; oy <= RADIUS; ++oy) {
        int yc = y + oy;
        yc = yc < 0 ? 0 : (yc > H_DIM - 1 ? H_DIM - 1 : yc);
#pragma unroll
        for (int ox = -RADIUS; ox <= RADIUS; ++ox) {
            int xc = x + ox;
            xc = xc < 0 ? 0 : (xc > W_DIM - 1 ? W_DIM - 1 : xc);
            gmapB[yc * W_DIM + xc] = g;   // gene 0 => zero embedding row
        }
    }
}

// ---------------------------------------------------------------------------
// Pass 2: out[b, :, p] = onehot(gmap[b,p]) @ weight  via v_wmma_f32_16x16x32_f16
// D = A(16ch x 32gene, f16 weight^T, preloaded) x B(32gene x 16px one-hot).
// Each wave: 4 pixel tiles x 4 channel tiles = 16 WMMAs, 16 KB of output.
// A-tiles loaded from LDS once per wave; stores use one vaddr + imm offsets.
// ---------------------------------------------------------------------------
__global__ void wmma_gather(const int* __restrict__ gmap,
                            const float* __restrict__ weight,
                            float* __restrict__ out,
                            int nGenes) {
    __shared__ _Float16 whT[EMBED * NGENE_PAD];   // [channel][gene], 4 KB

    // Cooperative fill: weight^T in f16, gene axis padded with zeros to 32.
    for (int idx = threadIdx.x; idx < EMBED * NGENE_PAD; idx += blockDim.x) {
        int ch = idx >> 5;
        int g  = idx & 31;
        whT[idx] = (g < nGenes) ? (_Float16)weight[g * EMBED + ch]
                                : (_Float16)0.0f;
    }
    __syncthreads();

    const int lane = threadIdx.x & 31;
    const int half = lane >> 4;                   // selects K block (B) / M block (D)
    const int ln   = lane & 15;

    const int b    = blockIdx.y;                  // scalar batch index
    const int wave = blockIdx.x * (blockDim.x >> 5) + (threadIdx.x >> 5);
    const int p0   = wave * (16 * PIX_TILES);     // first pixel of this wave

    const int*   gmapB = gmap + (size_t)b * HW;
    float*       outB  = out  + (size_t)b * EMBED * HW;

    // Preload all 4 channel A-tiles (weight^T) once; reused for every px tile.
    // 16-bit A 16x32 layout: a[0..7] = K 8*half..+7, a[8..15] = K 16+8*half..+7
    union { v16h v; v8h h[2]; } au[4];
#pragma unroll
    for (int m = 0; m < 4; ++m) {
        int ch = (m << 4) + ln;                   // A: M = channel (per lane)
        au[m].h[0] = *(const v8h*)&whT[ch * NGENE_PAD + (half << 3)];
        au[m].h[1] = *(const v8h*)&whT[ch * NGENE_PAD + 16 + (half << 3)];
    }

#pragma unroll
    for (int t = 0; t < PIX_TILES; ++t) {
        int px = p0 + (t << 4) + ln;              // this lane's pixel (B: N = ln)
        int g  = gmapB[px];

        // B one-hot (32x16, K x N): b[i] <-> K = 8*half + i (i<8), 16 + 8*half + i-8 (i>=8), N = ln.
        v16h bm;
#pragma unroll
        for (int i = 0; i < 16; ++i)
            bm[i] = (g == (i + (half << 3) + ((i >> 3) << 3))) ? (_Float16)1.0f : (_Float16)0.0f;

#pragma unroll
        for (int m = 0; m < 4; ++m) {
            v8f c = {};
            c = __builtin_amdgcn_wmma_f32_16x16x32_f16(
                    /*neg_a=*/false, au[m].v, /*neg_b=*/false, bm,
                    /*c_mod=*/(short)0, c, /*reuse_a=*/false, /*reuse_b=*/false);
            // D layout: VGPR r -> channel m*16 + r + 8*half, pixel px.
            // HW is compile-time: r*HW*4 < 2^23 -> immediate store offsets.
            float* o = outB + (((m << 4) + (half << 3)) * HW + px);
#pragma unroll
            for (int r = 0; r < 8; ++r)
                o[r * HW] = c[r];
        }
    }
}

extern "C" void kernel_launch(void* const* d_in, const int* in_sizes, int n_in,
                              void* d_out, int out_size, void* d_ws, size_t ws_size,
                              hipStream_t stream) {
    const int*   list_gene   = (const int*)d_in[0];   // [B, N]
    const int*   array_coord = (const int*)d_in[1];   // [B, N, 2]
    const float* weight      = (const float*)d_in[2]; // [nGenes, 64]
    // H, W arrive as device scalars; the reference fixes them at 512 and we
    // cannot read device memory during graph capture -> compile-time values.

    const int nGenes = in_sizes[2] / EMBED;           // 17
    const int B      = out_size / (EMBED * HW);       // 4
    const int N      = in_sizes[0] / B;               // 100000

    int* gmap = (int*)d_ws;                           // B*HW ints = 4 MB scratch

    zero_gmap<<<256, 256, 0, stream>>>((int4*)gmap, (B * HW) >> 2);

    dim3 sgrid((N + 255) / 256, B);
    scatter_genes<<<sgrid, 256, 0, stream>>>(list_gene, array_coord, gmap, N);

    // Each wave: 64 pixels; 8 waves/block -> 512 pixels/block.
    int wavesPerBatch = HW / (16 * PIX_TILES);        // 4096
    dim3 ggrid(wavesPerBatch / 8, B);                 // 512 x B, exact
    wmma_gather<<<ggrid, 256, 0, stream>>>(gmap, weight, (float*)d_out, nGenes);
}
